// CausalSelfAttention_adaptive_weight_68350109549022
// MI455X (gfx1250) — compile-verified, matrix operand order checked by reading, not run on hardware
//
#include <hip/hip_runtime.h>

// CDNA5 (gfx1250) wave32 WMMA implementation of adaptive-weight causal self-attention.
// D = A x B + C with v_wmma_f32_16x16x32_bf16 (f32 accumulate), register double-buffered GEMMs.

typedef __attribute__((ext_vector_type(16))) __bf16 v16bf;
typedef __attribute__((ext_vector_type(8)))  __bf16 v8bf;
typedef __attribute__((ext_vector_type(8)))  float  v8f;
typedef __attribute__((ext_vector_type(4)))  float  v4f;

#define BB 8
#define TT 1024
#define CC 1024
#define HH 16
#define HD 64

#if __has_builtin(__builtin_amdgcn_global_load_async_to_lds_b32)
#define HAS_ASYNC_LDS 1
typedef __attribute__((address_space(1))) int gint;   // AS1 ("__device__") int*
typedef __attribute__((address_space(3))) int lint;   // AS3 (LDS) int*
#else
#define HAS_ASYNC_LDS 0
#endif

static __device__ __forceinline__ v16bf cat8(v8bf lo, v8bf hi) {
  return __builtin_shufflevector(lo, hi, 0,1,2,3,4,5,6,7,8,9,10,11,12,13,14,15);
}

static __device__ __forceinline__ v8f wmma_bf16(v16bf a, v16bf b, v8f c) {
  // 8 args: (neg_a, A, neg_b, B, c_mod, C, reuse_a, reuse_b)
  return __builtin_amdgcn_wmma_f32_16x16x32_bf16(false, a, false, b, (short)0, c, false, false);
}

// Fragment load: two contiguous 16B chunks selected by lane half (ISA 16-bit layout).
static __device__ __forceinline__ v16bf frag16(const __bf16* p, int c0, int c1) {
  return cat8(*(const v8bf*)(p + c0), *(const v8bf*)(p + c1));
}

// ---------------- elementwise converts ----------------

__global__ void cvt_f32_to_bf16(const float* __restrict__ in, __bf16* __restrict__ out, int n) {
  int i = blockIdx.x * blockDim.x + threadIdx.x;
  if (i < n) out[i] = (__bf16)in[i];
}

// wt[n*C + k] = (bf16) w[k*C + n]   (C x C transpose; write-coalesced over k)
__global__ void transpose_to_bf16(const float* __restrict__ w, __bf16* __restrict__ wt) {
  int i = blockIdx.x * blockDim.x + threadIdx.x;
  int n = i >> 10, k = i & (CC - 1);
  wt[i] = (__bf16)w[k * CC + n];
}

// ---------------- GEMM1: qkv = x @ Wv + bv ; also v (row-weighted) transposed per head ----------------
// Per-wave 32x64 tile (2 A frags x 4 B frags, 8 WMMAs/K-step), register double-buffered.
// Block = 128 threads (4 waves) -> 128x64 output tile.

__global__ __launch_bounds__(128)
void gemm_qkv(const __bf16* __restrict__ xb, const __bf16* __restrict__ wvt,
              const float* __restrict__ bv, const float* __restrict__ rw,
              __bf16* __restrict__ qkvb, __bf16* __restrict__ vT) {
  __shared__ float biasLds[64];
  const int lane = threadIdx.x & 31;
  const int wave = threadIdx.x >> 5;
  const int half = lane >> 4;
  const int m    = lane & 15;
  const int row0 = blockIdx.y * 128 + wave * 32;
  const int n0   = blockIdx.x * 64;
  const int c0   = half * 8;        // ISA 16-bit fragment K sub-chunk 0
  const int c1   = 16 + half * 8;   // sub-chunk 1

  // stage bias for this block's 64 columns through the CDNA5 async-to-LDS path
#if HAS_ASYNC_LDS
  if (threadIdx.x < 64) {
    gint* gp = (gint*)(unsigned long long)(bv + n0 + threadIdx.x);
    lint* lp = (lint*)(unsigned)(unsigned long long)(&biasLds[threadIdx.x]);
    __builtin_amdgcn_global_load_async_to_lds_b32(gp, lp, 0, 0);
  }
#else
  if (threadIdx.x < 64) biasLds[threadIdx.x] = bv[n0 + threadIdx.x];
#endif

  const __bf16* arow0 = xb + (size_t)(row0 + m) * CC;
  const __bf16* arow1 = xb + (size_t)(row0 + 16 + m) * CC;
  const __bf16* brow[4];
#pragma unroll
  for (int f = 0; f < 4; ++f) brow[f] = wvt + (size_t)(n0 + f * 16 + m) * CC;

  v8f acc[2][4] = {};
  v16bf a[2], bfrag[4];
  a[0] = frag16(arow0, c0, c1);
  a[1] = frag16(arow1, c0, c1);
#pragma unroll
  for (int f = 0; f < 4; ++f) bfrag[f] = frag16(brow[f], c0, c1);

#pragma unroll 2
  for (int k0 = 0; k0 < CC; k0 += 32) {
    const int kn = (k0 + 32) & (CC - 1);     // wraps to 0 on last iter (dead loads, valid addr)
    v16bf an[2], bn[4];
    an[0] = frag16(arow0 + kn, c0, c1);
    an[1] = frag16(arow1 + kn, c0, c1);
#pragma unroll
    for (int f = 0; f < 4; ++f) bn[f] = frag16(brow[f] + kn, c0, c1);
    __builtin_prefetch(arow0 + k0 + 512, 0, 1);  // global_prefetch_b8, ~1KB ahead
    __builtin_prefetch(arow1 + k0 + 512, 0, 1);
#pragma unroll
    for (int f = 0; f < 4; ++f) {
      acc[0][f] = wmma_bf16(a[0], bfrag[f], acc[0][f]);
      acc[1][f] = wmma_bf16(a[1], bfrag[f], acc[1][f]);
    }
    a[0] = an[0]; a[1] = an[1];
#pragma unroll
    for (int f = 0; f < 4; ++f) bfrag[f] = bn[f];
  }

#if HAS_ASYNC_LDS
  asm volatile("s_wait_asynccnt 0" ::: "memory");
#endif
  __syncthreads();

#pragma unroll
  for (int i = 0; i < 2; ++i) {
#pragma unroll
    for (int f = 0; f < 4; ++f) {
      const int n = n0 + f * 16 + m;
      const float bias = biasLds[f * 16 + m];
      const int h = n >> 6, d = n & (HD - 1);
#pragma unroll
      for (int r = 0; r < 8; ++r) {
        const int row = row0 + i * 16 + r + half * 8;   // C-layout: M = r + half*8
        const float v = acc[i][f][r] + bias;
        qkvb[(size_t)row * CC + n] = (__bf16)v;
        const int bidx = row >> 10, t = row & (TT - 1);
        vT[(((size_t)bidx * HH + h) * HD + d) * TT + t] = (__bf16)(v * rw[row]);
      }
    }
  }
}

// ---------------- Flash attention: one wave per (b, h, 16-query tile) ----------------

__global__ __launch_bounds__(128)
void attn(const __bf16* __restrict__ qkvb, const __bf16* __restrict__ vT,
          const float* __restrict__ rw, float* __restrict__ y) {
  __shared__ __bf16 plds[4][16 * 32];          // per-wave P-tile for C->A relayout
  const int lane = threadIdx.x & 31;
  const int wave = threadIdx.x >> 5;
  const int half = lane >> 4;
  const int m    = lane & 15;
  const int wid  = blockIdx.x * 4 + wave;      // 8192 waves total
  const int qt   = wid & 63;                   // T/16
  const int h    = (wid >> 6) & (HH - 1);
  const int b    = wid >> 10;
  const int q0   = qt * 16;
  const int c0   = half * 8, c1 = 16 + half * 8;

  // Q fragments (hd=64 -> two 16x32 A fragments)
  const __bf16* qrow = qkvb + (size_t)(b * TT + q0 + m) * CC + h * HD;
  const v16bf qa0 = frag16(qrow, c0, c1);
  const v16bf qa1 = frag16(qrow + 32, c0, c1);

  v8f yacc[4] = {};
  float mrow[8], lrow[8];
#pragma unroll
  for (int r = 0; r < 8; ++r) { mrow[r] = -1e30f; lrow[r] = 0.f; }

  const float scale = 0.125f;                  // 1/sqrt(64)
  const int nchunks = q0 / 32 + 1;             // causal: keys up to q0+15 covered
  __bf16* pb = plds[wave];

  for (int kc = 0; kc < nchunks; ++kc) {
    const int kb = kc * 32;
    // ---- S = Q @ K^T for 32 keys (two 16x16 tiles) ----
    v8f s0 = {}, s1 = {};
    const __bf16* krow0 = qkvb + (size_t)(b * TT + kb + m) * CC + h * HD;
    {
      v16bf b00 = frag16(krow0, c0, c1);
      v16bf b01 = frag16(krow0 + 32, c0, c1);
      s0 = wmma_bf16(qa0, b00, s0);
      s0 = wmma_bf16(qa1, b01, s0);
      const __bf16* krow1 = krow0 + (size_t)16 * CC;
      v16bf b10 = frag16(krow1, c0, c1);
      v16bf b11 = frag16(krow1 + 32, c0, c1);
      s1 = wmma_bf16(qa0, b10, s1);
      s1 = wmma_bf16(qa1, b11, s1);
    }
    // ---- adaptive key weighting + scale + causal mask ----
    const int key0 = kb + m, key1 = kb + 16 + m;
    const float rw0 = rw[b * TT + key0] * scale;
    const float rw1 = rw[b * TT + key1] * scale;
    float mx[8];
#pragma unroll
    for (int r = 0; r < 8; ++r) {
      const int q = q0 + r + half * 8;
      float a0 = s0[r] * rw0; if (key0 > q) a0 = -1e30f;
      float a1 = s1[r] * rw1; if (key1 > q) a1 = -1e30f;
      s0[r] = a0; s1[r] = a1;
      mx[r] = fmaxf(a0, a1);
    }
    // row-max across the 16 lanes of each half (wave32 shuffles)
#pragma unroll
    for (int off = 1; off < 16; off <<= 1)
#pragma unroll
      for (int r = 0; r < 8; ++r)
        mx[r] = fmaxf(mx[r], __shfl_xor(mx[r], off, 32));

    float alpha[8], sum[8];
#pragma unroll
    for (int r = 0; r < 8; ++r) {
      const float mn = fmaxf(mrow[r], mx[r]);
      alpha[r] = __expf(mrow[r] - mn);
      mrow[r]  = mn;
      const float p0 = __expf(s0[r] - mn);
      const float p1 = __expf(s1[r] - mn);
      sum[r] = p0 + p1;
      // park P in LDS in C-layout: P[M=r+half*8][key-in-chunk]
      pb[(r + half * 8) * 32 + m]      = (__bf16)p0;
      pb[(r + half * 8) * 32 + 16 + m] = (__bf16)p1;
    }
#pragma unroll
    for (int off = 1; off < 16; off <<= 1)
#pragma unroll
      for (int r = 0; r < 8; ++r)
        sum[r] += __shfl_xor(sum[r], off, 32);
#pragma unroll
    for (int r = 0; r < 8; ++r) lrow[r] = lrow[r] * alpha[r] + sum[r];
#pragma unroll
    for (int f = 0; f < 4; ++f)
#pragma unroll
      for (int r = 0; r < 8; ++r)
        yacc[f][r] *= alpha[r];

    asm volatile("s_wait_dscnt 0" ::: "memory");  // LDS P-tile visible to all lanes of wave

    // reload P as a 16x32 A fragment (per-lane row mA=m, K sub-chunks by half)
    const v16bf pa = cat8(*(const v8bf*)(pb + m * 32 + half * 8),
                          *(const v8bf*)(pb + m * 32 + 16 + half * 8));
    // ---- Y += P @ V (V pre-transposed: contiguous per-lane b128 loads) ----
#pragma unroll
    for (int f = 0; f < 4; ++f) {
      const __bf16* vrow = vT + (((size_t)b * HH + h) * HD + f * 16 + m) * TT + kb;
      v16bf vb = frag16(vrow, c0, c1);
      yacc[f] = wmma_bf16(pa, vb, yacc[f]);
    }
  }
  // ---- normalize and write y (fp32, row-major [B*T][C]) ----
#pragma unroll
  for (int f = 0; f < 4; ++f)
#pragma unroll
    for (int r = 0; r < 8; ++r) {
      const int row = b * TT + q0 + r + half * 8;
      const int col = h * HD + f * 16 + m;
      y[(size_t)row * CC + col] = yacc[f][r] / lrow[r];
    }
}

// ---------------- GEMM2: out = y @ Wproj + bproj (A fp32 -> bf16 on the fly) ----------------
// Same 32x64 per-wave tile; next A tile is kept raw (v4f) so its load-wait lands
// behind the current iteration's WMMAs; converted at the top of the next iteration.

__global__ __launch_bounds__(128)
void gemm_proj(const float* __restrict__ y, const __bf16* __restrict__ wpt,
               const float* __restrict__ bp, float* __restrict__ out) {
  const int lane = threadIdx.x & 31;
  const int wave = threadIdx.x >> 5;
  const int half = lane >> 4;
  const int m    = lane & 15;
  const int row0 = blockIdx.y * 128 + wave * 32;
  const int n0   = blockIdx.x * 64;
  const int c0   = half * 8, c1 = 16 + half * 8;

  const float* arow[2] = { y + (size_t)(row0 + m) * CC, y + (size_t)(row0 + 16 + m) * CC };
  const __bf16* brow[4];
#pragma unroll
  for (int f = 0; f < 4; ++f) brow[f] = wpt + (size_t)(n0 + f * 16 + m) * CC;

  v8f acc[2][4] = {};
  v4f araw[2][4];
  v16bf bfrag[4];
#pragma unroll
  for (int i = 0; i < 2; ++i) {
    araw[i][0] = *(const v4f*)(arow[i] + c0);
    araw[i][1] = *(const v4f*)(arow[i] + c0 + 4);
    araw[i][2] = *(const v4f*)(arow[i] + c1);
    araw[i][3] = *(const v4f*)(arow[i] + c1 + 4);
  }
#pragma unroll
  for (int f = 0; f < 4; ++f) bfrag[f] = frag16(brow[f], c0, c1);

#pragma unroll 2
  for (int k0 = 0; k0 < CC; k0 += 32) {
    const int kn = (k0 + 32) & (CC - 1);
    v4f anraw[2][4];
    v16bf bn[4];
#pragma unroll
    for (int i = 0; i < 2; ++i) {
      anraw[i][0] = *(const v4f*)(arow[i] + kn + c0);
      anraw[i][1] = *(const v4f*)(arow[i] + kn + c0 + 4);
      anraw[i][2] = *(const v4f*)(arow[i] + kn + c1);
      anraw[i][3] = *(const v4f*)(arow[i] + kn + c1 + 4);
    }
#pragma unroll
    for (int f = 0; f < 4; ++f) bn[f] = frag16(brow[f] + kn, c0, c1);
    __builtin_prefetch(arow[0] + k0 + 256, 0, 1);
    __builtin_prefetch(arow[1] + k0 + 256, 0, 1);

    v16bf a[2];
#pragma unroll
    for (int i = 0; i < 2; ++i)
#pragma unroll
      for (int j = 0; j < 4; ++j)
#pragma unroll
        for (int e = 0; e < 4; ++e)
          a[i][j * 4 + e] = (__bf16)araw[i][j][e];

#pragma unroll
    for (int f = 0; f < 4; ++f) {
      acc[0][f] = wmma_bf16(a[0], bfrag[f], acc[0][f]);
      acc[1][f] = wmma_bf16(a[1], bfrag[f], acc[1][f]);
    }
#pragma unroll
    for (int i = 0; i < 2; ++i)
#pragma unroll
      for (int j = 0; j < 4; ++j) araw[i][j] = anraw[i][j];
#pragma unroll
    for (int f = 0; f < 4; ++f) bfrag[f] = bn[f];
  }

#pragma unroll
  for (int i = 0; i < 2; ++i)
#pragma unroll
    for (int f = 0; f < 4; ++f) {
      const int n = n0 + f * 16 + m;
      const float bias = bp[n];
#pragma unroll
      for (int r = 0; r < 8; ++r) {
        const int row = row0 + i * 16 + r + half * 8;
        out[(size_t)row * CC + n] = acc[i][f][r] + bias;
      }
    }
}

// ---------------- host launcher ----------------

extern "C" void kernel_launch(void* const* d_in, const int* in_sizes, int n_in,
                              void* d_out, int out_size, void* d_ws, size_t ws_size,
                              hipStream_t stream) {
  (void)in_sizes; (void)n_in; (void)out_size; (void)ws_size;
  const float* x  = (const float*)d_in[0];
  const float* rw = (const float*)d_in[1];
  const float* Wv = (const float*)d_in[2];
  const float* bv = (const float*)d_in[3];
  const float* Wp = (const float*)d_in[4];
  const float* bp = (const float*)d_in[5];
  float* out = (float*)d_out;

  char* ws = (char*)d_ws;
  size_t off = 0;
  __bf16* xb   = (__bf16*)(ws + off); off += (size_t)BB * TT * CC * 2;  // x in bf16
  __bf16* wvt  = (__bf16*)(ws + off); off += (size_t)CC * CC * 2;       // Wv^T bf16
  __bf16* wpt  = (__bf16*)(ws + off); off += (size_t)CC * CC * 2;       // Wproj^T bf16
  __bf16* qkvb = (__bf16*)(ws + off); off += (size_t)BB * TT * CC * 2;  // qkv bf16 (Q,K)
  __bf16* vT   = (__bf16*)(ws + off); off += (size_t)BB * TT * CC * 2;  // v per-head transposed
  float*  yws  = (float*)(ws + off);  off += (size_t)BB * TT * CC * 4;  // attention out fp32

  const int nx = BB * TT * CC;
  cvt_f32_to_bf16<<<nx / 256, 256, 0, stream>>>(x, xb, nx);
  transpose_to_bf16<<<(CC * CC) / 256, 256, 0, stream>>>(Wv, wvt);
  transpose_to_bf16<<<(CC * CC) / 256, 256, 0, stream>>>(Wp, wpt);
  gemm_qkv<<<dim3(16, 64), 128, 0, stream>>>(xb, wvt, bv, rw, qkvb, vT);
  attn<<<2048, 128, 0, stream>>>(qkvb, vT, rw, yws);
  gemm_proj<<<dim3(16, 64), 128, 0, stream>>>(yws, wpt, bp, out);
}
